// MultiHeadAttention_27247272526033
// MI455X (gfx1250) — matrix operand order repaired and checked by reading, not compiled, not run on hardware
//
#include <hip/hip_runtime.h>
#include <hip/hip_bf16.h>

typedef __attribute__((ext_vector_type(16))) __bf16 v16bf;
typedef __attribute__((ext_vector_type(8)))  __bf16 v8bf;
typedef __attribute__((ext_vector_type(8)))  float  v8f;

#define B_  2
#define L_  2048
#define D_  1024
#define H_  16
#define DK_ 64

static __device__ __forceinline__ __bf16 f2bf(float f) {
  unsigned u = __builtin_bit_cast(unsigned, f);
  unsigned r = u + 0x7FFFu + ((u >> 16) & 1u);
  unsigned short h = (unsigned short)(r >> 16);
  return __builtin_bit_cast(__bf16, h);
}

__global__ void __launch_bounds__(256) cast_f32_to_bf16(
    const float* __restrict__ in, __bf16* __restrict__ out, int n) {
  for (int i = blockIdx.x * blockDim.x + threadIdx.x; i < n;
       i += gridDim.x * blockDim.x)
    out[i] = f2bf(in[i]);
}

// A fragment (16x32 bf16): lane (mr, g) holds row mr, K=[16g,16g+16)
// -- two 16-byte chunks.
static __device__ __forceinline__ v16bf load_a_frag(const __bf16* arow,
                                                    int k0, int g) {
  v8bf c0 = *(const v8bf*)(arow + k0 + 16 * g);
  v8bf c1 = *(const v8bf*)(arow + k0 + 16 * g + 8);
  v16bf af;
#pragma unroll
  for (int i = 0; i < 8; ++i) { af[i] = c0[i]; af[8 + i] = c1[i]; }
  return af;
}

// Wave computes a 32(M) x 64(N) tile of A(M x 1024) * W^T, W stored (N x 1024)
// row-major. Each B fragment feeds two WMMAs (one per 16-row M subtile), so
// fragment-load latency amortizes over pairs of matrix ops; B loads for the
// next subtile issue over the current WMMA pair.
static __device__ __forceinline__ void gemm32x64_nt(
    const __bf16* __restrict__ A, const __bf16* __restrict__ W,
    int m0, int n0, v8f acc[2][4]) {
  const int lane = threadIdx.x & 31;
  const int mr = lane & 15;
  const int g  = lane >> 4;
  const __bf16* arow0 = A + (size_t)(m0 + mr) * D_;
  const __bf16* arow1 = arow0 + (size_t)16 * D_;
  const __bf16* wrow  = W + (size_t)(n0 + mr) * D_;

#pragma unroll 2
  for (int k0 = 0; k0 < D_; k0 += 32) {
    // Prefetch ahead (speculative, dropped if OOB).
    __builtin_prefetch(arow0 + k0 + 256, 0, 1);
    __builtin_prefetch(wrow + k0 + 256, 0, 1);
    v16bf a0 = load_a_frag(arow0, k0, g);
    v16bf a1 = load_a_frag(arow1, k0, g);
#pragma unroll
    for (int t = 0; t < 4; ++t) {
      v16bf bt = *(const v16bf*)(wrow + (size_t)t * 16 * D_ + k0 + 16 * g);
      acc[0][t] = __builtin_amdgcn_wmma_f32_16x16x32_bf16(
          false, a0, false, bt, (short)0, acc[0][t], false, false);
      acc[1][t] = __builtin_amdgcn_wmma_f32_16x16x32_bf16(
          false, a1, false, bt, (short)0, acc[1][t], false, false);
    }
  }
}

// Q/K projection: out layout (B, H, L, DK), bf16.
__global__ void __launch_bounds__(256) proj_qk_kernel(
    const __bf16* __restrict__ X, const __bf16* __restrict__ W,
    __bf16* __restrict__ out) {
  const int wave = threadIdx.x >> 5;
  const int lane = threadIdx.x & 31;
  const int mr = lane & 15, g = lane >> 4;
  const int m0 = (blockIdx.x * 8 + wave) * 32;
  const int n0 = blockIdx.y * 64;
  v8f acc[2][4] = {};
  gemm32x64_nt(X, W, m0, n0, acc);
#pragma unroll
  for (int u = 0; u < 2; ++u) {
#pragma unroll
    for (int t = 0; t < 4; ++t) {
      int n = n0 + t * 16 + mr;
      int h = n >> 6, d = n & (DK_ - 1);
#pragma unroll
      for (int r = 0; r < 8; ++r) {
        int m = m0 + 16 * u + 8 * g + r;
        int b = m >> 11, l = m & (L_ - 1);
        out[(((size_t)b * H_ + h) * L_ + l) * DK_ + d] = f2bf(acc[u][t][r]);
      }
    }
  }
}

// V projection, stored transposed per head: Vt layout (B, H, DK, L), bf16.
// C-layout gives each lane 8 contiguous M values -> one packed b128 store.
__global__ void __launch_bounds__(256) proj_v_kernel(
    const __bf16* __restrict__ X, const __bf16* __restrict__ W,
    __bf16* __restrict__ Vt) {
  const int wave = threadIdx.x >> 5;
  const int lane = threadIdx.x & 31;
  const int mr = lane & 15, g = lane >> 4;
  const int m0 = (blockIdx.x * 8 + wave) * 32;
  const int n0 = blockIdx.y * 64;
  v8f acc[2][4] = {};
  gemm32x64_nt(X, W, m0, n0, acc);
#pragma unroll
  for (int u = 0; u < 2; ++u) {
    const int mbase = m0 + 16 * u + 8 * g;
    const int b = mbase >> 11, l0 = mbase & (L_ - 1);
#pragma unroll
    for (int t = 0; t < 4; ++t) {
      int n = n0 + t * 16 + mr;
      int h = n >> 6, d = n & (DK_ - 1);
      v8bf pk;
#pragma unroll
      for (int r = 0; r < 8; ++r) pk[r] = f2bf(acc[u][t][r]);
      *(v8bf*)(Vt + (((size_t)b * H_ + h) * DK_ + d) * L_ + l0) = pk;
    }
  }
}

// Fused flash attention: one wave per 16-query tile, 32 keys per iteration.
// Pipelining: V fragments load at iteration start (consumed after softmax),
// next iteration's K fragments load right after QK^T issues.
__global__ void __launch_bounds__(256) attention_kernel(
    const __bf16* __restrict__ Qp, const __bf16* __restrict__ Kp,
    const __bf16* __restrict__ Vt, const int* __restrict__ mask,
    __bf16* __restrict__ ctxb) {
  __shared__ __bf16 Pl[8][16 * 32];   // per-wave P tile, row-major 16x32
  const int wave = threadIdx.x >> 5;
  const int lane = threadIdx.x & 31;
  const int mr = lane & 15, g = lane >> 4;
  const int b = blockIdx.z, h = blockIdx.y;
  const int q0 = (blockIdx.x * 8 + wave) * 16;
  const size_t bh = (size_t)b * H_ + h;

  const __bf16* kbase  = Kp + bh * L_ * DK_;
  const __bf16* vbase  = Vt + (bh * DK_ + mr) * L_ + 16 * g;
  const int*    mbase_ = mask + b * L_ + mr;

  // Q A-fragments for d-chunks [0,32) and [32,64)
  const __bf16* qrow = Qp + (bh * L_ + q0 + mr) * DK_;
  v16bf qa[2];
#pragma unroll
  for (int c = 0; c < 2; ++c)
    qa[c] = load_a_frag(qrow, c * 32, g);

  float rmax[8], rsum[8];
#pragma unroll
  for (int r = 0; r < 8; ++r) { rmax[r] = -3.0e38f; rsum[r] = 0.f; }
  v8f ctx[4] = {};

  // Preload first K tile fragments.
  v16bf kf[2][2];
#pragma unroll
  for (int t = 0; t < 2; ++t) {
    const __bf16* krow = kbase + (size_t)(t * 16 + mr) * DK_;
#pragma unroll
    for (int c = 0; c < 2; ++c)
      kf[t][c] = *(const v16bf*)(krow + c * 32 + 16 * g);
  }

  for (int j = 0; j < L_ / 32; ++j) {
    const int k0 = j * 32;
    // V fragments for this iteration (consumed only after softmax).
    v16bf vf[4];
#pragma unroll
    for (int c = 0; c < 4; ++c)
      vf[c] = *(const v16bf*)(vbase + (size_t)c * 16 * L_ + k0);
    // Mask for this lane's two columns.
    const int mv0 = mbase_[k0];
    const int mv1 = mbase_[k0 + 16];

    // S = Q * K^T for 32 keys (two 16x16 C tiles, K-dim = d in 2 chunks)
    v8f s[2] = {};
#pragma unroll
    for (int t = 0; t < 2; ++t)
#pragma unroll
      for (int c = 0; c < 2; ++c)
        s[t] = __builtin_amdgcn_wmma_f32_16x16x32_bf16(
            false, qa[c], false, kf[t][c], (short)0, s[t], false, false);

    // Issue next iteration's K loads (clamped, branch-free).
    const int k0n = (k0 + 32 < L_) ? k0 + 32 : k0;
    v16bf kfn[2][2];
#pragma unroll
    for (int t = 0; t < 2; ++t) {
      const __bf16* krow = kbase + (size_t)(k0n + t * 16 + mr) * DK_;
#pragma unroll
      for (int c = 0; c < 2; ++c)
        kfn[t][c] = *(const v16bf*)(krow + c * 32 + 16 * g);
    }

    // Online softmax; row m = r + 8g lives in this lane's 16-lane half.
#pragma unroll
    for (int r = 0; r < 8; ++r) {
      float s0 = mv0 ? s[0][r] * 0.125f : -10000.f;  // /sqrt(64)
      float s1 = mv1 ? s[1][r] * 0.125f : -10000.f;
      float m2 = fmaxf(s0, s1);
#pragma unroll
      for (int off = 1; off < 16; off <<= 1)
        m2 = fmaxf(m2, __shfl_xor(m2, off, 32));
      float mn = fmaxf(rmax[r], m2);
      float alpha = __expf(rmax[r] - mn);
      rmax[r] = mn;
      float p0 = __expf(s0 - mn);
      float p1 = __expf(s1 - mn);
      float ps = p0 + p1;
#pragma unroll
      for (int off = 1; off < 16; off <<= 1) ps += __shfl_xor(ps, off, 32);
      rsum[r] = rsum[r] * alpha + ps;
#pragma unroll
      for (int c = 0; c < 4; ++c) ctx[c][r] *= alpha;
      Pl[wave][(r + 8 * g) * 32 + mr] = f2bf(p0);
      Pl[wave][(r + 8 * g) * 32 + 16 + mr] = f2bf(p1);
    }
    __syncthreads();               // uniform across all 8 waves every iter
    // Re-read P in A-fragment layout (C-layout -> A-layout via LDS)
    v16bf pa = load_a_frag(&Pl[wave][0] + (size_t)mr * 32, 0, g);
    __syncthreads();
    // ctx += P * V  (B operand contiguous thanks to pre-transposed Vt)
#pragma unroll
    for (int c = 0; c < 4; ++c)
      ctx[c] = __builtin_amdgcn_wmma_f32_16x16x32_bf16(
          false, pa, false, vf[c], (short)0, ctx[c], false, false);
    // Rotate K double buffer.
#pragma unroll
    for (int t = 0; t < 2; ++t)
#pragma unroll
      for (int c = 0; c < 2; ++c) kf[t][c] = kfn[t][c];
  }
  // Normalize and store ctx in (B, L, D) layout
#pragma unroll
  for (int c = 0; c < 4; ++c) {
    int d = h * DK_ + c * 16 + mr;
#pragma unroll
    for (int r = 0; r < 8; ++r) {
      int l = q0 + 8 * g + r;
      ctxb[((size_t)b * L_ + l) * D_ + d] = f2bf(ctx[c][r] / rsum[r]);
    }
  }
}

// Output projection: ctx @ w_o^T + b_o, f32 output.
__global__ void __launch_bounds__(256) out_proj_kernel(
    const __bf16* __restrict__ Ctx, const __bf16* __restrict__ W,
    const float* __restrict__ bo, float* __restrict__ out) {
  const int wave = threadIdx.x >> 5;
  const int lane = threadIdx.x & 31;
  const int mr = lane & 15, g = lane >> 4;
  const int m0 = (blockIdx.x * 8 + wave) * 32;
  const int n0 = blockIdx.y * 64;
  v8f acc[2][4] = {};
  gemm32x64_nt(Ctx, W, m0, n0, acc);
#pragma unroll
  for (int u = 0; u < 2; ++u) {
#pragma unroll
    for (int t = 0; t < 4; ++t) {
      int n = n0 + t * 16 + mr;
      float bias = bo[n];
#pragma unroll
      for (int r = 0; r < 8; ++r) {
        int m = m0 + 16 * u + 8 * g + r;
        out[(size_t)m * D_ + n] = acc[u][t][r] + bias;
      }
    }
  }
}

extern "C" void kernel_launch(void* const* d_in, const int* in_sizes, int n_in,
                              void* d_out, int out_size, void* d_ws, size_t ws_size,
                              hipStream_t stream) {
  const float* q    = (const float*)d_in[0];
  const float* k    = (const float*)d_in[1];
  const float* v    = (const float*)d_in[2];
  const int*   mask = (const int*)d_in[3];
  const float* wq   = (const float*)d_in[4];
  const float* wk   = (const float*)d_in[5];
  const float* wv   = (const float*)d_in[6];
  const float* wo   = (const float*)d_in[7];
  const float* bo   = (const float*)d_in[8];
  float* out = (float*)d_out;
  (void)in_sizes; (void)n_in; (void)out_size; (void)ws_size;

  const size_t BLD = (size_t)B_ * L_ * D_;   // 4,194,304
  const size_t DD  = (size_t)D_ * D_;        // 1,048,576
  char* w = (char*)d_ws;
  auto take = [&](size_t bytes) {
    char* p = w;
    w += (bytes + 255) & ~(size_t)255;
    return p;
  };
  __bf16* qb  = (__bf16*)take(BLD * 2);
  __bf16* kb  = (__bf16*)take(BLD * 2);
  __bf16* vb  = (__bf16*)take(BLD * 2);
  __bf16* wqb = (__bf16*)take(DD * 2);
  __bf16* wkb = (__bf16*)take(DD * 2);
  __bf16* wvb = (__bf16*)take(DD * 2);
  __bf16* wob = (__bf16*)take(DD * 2);
  __bf16* Qp  = (__bf16*)take(BLD * 2);   // (B,H,L,DK)
  __bf16* Kp  = (__bf16*)take(BLD * 2);   // (B,H,L,DK)
  __bf16* Vt  = (__bf16*)take(BLD * 2);   // (B,H,DK,L)
  __bf16* Ctx = (__bf16*)take(BLD * 2);   // (B,L,D)

  cast_f32_to_bf16<<<1024, 256, 0, stream>>>(q,  qb,  (int)BLD);
  cast_f32_to_bf16<<<1024, 256, 0, stream>>>(k,  kb,  (int)BLD);
  cast_f32_to_bf16<<<1024, 256, 0, stream>>>(v,  vb,  (int)BLD);
  cast_f32_to_bf16<<<512,  256, 0, stream>>>(wq, wqb, (int)DD);
  cast_f32_to_bf16<<<512,  256, 0, stream>>>(wk, wkb, (int)DD);
  cast_f32_to_bf16<<<512,  256, 0, stream>>>(wv, wvb, (int)DD);
  cast_f32_to_bf16<<<512,  256, 0, stream>>>(wo, wob, (int)DD);

  dim3 gp(B_ * L_ / 256, D_ / 64);   // 16 x 16 blocks, 8 waves x 32 rows
  proj_qk_kernel<<<gp, 256, 0, stream>>>(qb, wqb, Qp);
  proj_qk_kernel<<<gp, 256, 0, stream>>>(kb, wkb, Kp);
  proj_v_kernel <<<gp, 256, 0, stream>>>(vb, wvb, Vt);

  dim3 ga(L_ / 128, H_, B_);         // 16 x 16 x 2 blocks, 8 waves each
  attention_kernel<<<ga, 256, 0, stream>>>(Qp, Kp, Vt, mask, Ctx);

  out_proj_kernel<<<gp, 256, 0, stream>>>(Ctx, wob, bo, out);
}
